// Custom_Layer_89644557402935
// MI455X (gfx1250) — compile-verified, matrix operand order checked by reading, not run on hardware
//
#include <hip/hip_runtime.h>
#include <stddef.h>
#include <stdint.h>

// ---------------------------------------------------------------------------
// Problem dimensions (hardcoded to the reference)
// ---------------------------------------------------------------------------
constexpr int Bsz = 8192;   // batch
constexpr int Tt  = 4;      // timesteps
constexpr int Dd  = 2048;   // input feature dim
constexpr int Hh  = 1024;   // hidden dim
constexpr int Uu  = 2048;   // output units

// GEMM tiling
constexpr int BM   = 128;
constexpr int BN   = 128;
constexpr int BK   = 32;
constexpr int LDSK = 40;            // padded LDS row stride (80B, 16B-aligned)
constexpr int ASZ  = BM * LDSK;     // one A buffer, elements
constexpr int BSZ  = BN * LDSK;     // one B buffer, elements

typedef __bf16 v16bf __attribute__((ext_vector_type(16)));
typedef __bf16 v8bf  __attribute__((ext_vector_type(8)));
typedef __bf16 v4bf  __attribute__((ext_vector_type(4)));
typedef float  v8f   __attribute__((ext_vector_type(8)));
typedef float  f32x4 __attribute__((ext_vector_type(4)));
typedef uint32_t u32x4 __attribute__((ext_vector_type(4)));
typedef int    v4i   __attribute__((ext_vector_type(4)));

// ---------------------------------------------------------------------------
// CDNA5 async global->LDS copy (GLOBAL_LOAD_ASYNC_TO_LDS_B128, ASYNCcnt)
// ---------------------------------------------------------------------------
#if defined(__has_builtin)
#  if __has_builtin(__builtin_amdgcn_global_load_async_to_lds_b128)
#    define HAVE_ASYNC_LDS 1
#  endif
#endif
#ifndef HAVE_ASYNC_LDS
#  define HAVE_ASYNC_LDS 0
#endif

typedef __attribute__((address_space(1))) v4i gv4i;   // global int4
typedef __attribute__((address_space(3))) v4i lv4i;   // LDS int4

#if HAVE_ASYNC_LDS
__device__ __forceinline__ void async_cp16(const __bf16* g, __bf16* l) {
  __builtin_amdgcn_global_load_async_to_lds_b128((gv4i*)g, (lv4i*)l, 0, 0);
}
__device__ __forceinline__ void wait_async0() {
#  if __has_builtin(__builtin_amdgcn_s_wait_asynccnt)
  __builtin_amdgcn_s_wait_asynccnt(0);
#  else
  asm volatile("s_wait_asynccnt 0" ::: "memory");
#  endif
}
#endif

// Load one 16x32 bf16 fragment (A-style layout) from a padded LDS tile.
// lane 0-15: row = row0+lane,    K chunks [0..8) and [16..24)
// lane16-31: row = row0+lane-16, K chunks [8..16) and [24..32)
__device__ __forceinline__ v16bf load_frag(const __bf16* lds, int row0, int lane) {
  const __bf16* base = lds + (row0 + (lane & 15)) * LDSK + ((lane >> 4) << 3);
  v8bf lo = *(const v8bf*)(base);
  v8bf hi = *(const v8bf*)(base + 16);
  return __builtin_shufflevector(lo, hi, 0, 1, 2, 3, 4, 5, 6, 7,
                                          8, 9, 10, 11, 12, 13, 14, 15);
}

// One BK=32 step of the 128x128 block: 8 waves in a 4x2 grid, each wave does
// a 32x64 sub-tile = 2x4 v_wmma_f32_16x16x32_bf16.
__device__ __forceinline__ void mma_block(const __bf16* lA, const __bf16* lB,
                                          int tid, v8f acc[2][4]) {
  int lane = tid & 31;
  int wave = tid >> 5;
  int wm = wave & 3;   // 0..3 -> M offset wm*32
  int wn = wave >> 2;  // 0..1 -> N offset wn*64
  v16bf a[2], b[4];
#pragma unroll
  for (int im = 0; im < 2; ++im) a[im] = load_frag(lA, wm * 32 + im * 16, lane);
#pragma unroll
  for (int in = 0; in < 4; ++in) b[in] = load_frag(lB, wn * 64 + in * 16, lane);
#pragma unroll
  for (int im = 0; im < 2; ++im)
#pragma unroll
    for (int in = 0; in < 4; ++in)
      acc[im][in] = __builtin_amdgcn_wmma_f32_16x16x32_bf16(
          false, a[im], false, b[in], (short)0, acc[im][in], false, false);
}

// Full K loop: A is [M][K] bf16 row-major, Bt is the weight pre-transposed to
// [N][K] bf16 row-major (so both operands use the same fragment path).
// sA/sB point at 2 concatenated LDS buffers (double-buffered on the async path).
__device__ __forceinline__ void gemm_tiles(const __bf16* __restrict__ A, int lda,
                                           const __bf16* __restrict__ Bt, int ldb,
                                           int K, int mBase, int nBase,
                                           __bf16* sA, __bf16* sB,
                                           v8f acc[2][4], int tid) {
  // Each thread owns two 16B chunks per 128x32 tile: (row r0, col c0) and
  // (row r0+64, col c0).
  const int r0 = tid >> 2;
  const int c0 = (tid & 3) << 3;
  const __bf16* gA0 = A + (size_t)(mBase + r0) * lda + c0;
  const __bf16* gA1 = gA0 + (size_t)64 * lda;
  const __bf16* gB0 = Bt + (size_t)(nBase + r0) * ldb + c0;
  const __bf16* gB1 = gB0 + (size_t)64 * ldb;
  __bf16* dA0 = sA + r0 * LDSK + c0;
  __bf16* dA1 = dA0 + 64 * LDSK;
  __bf16* dB0 = sB + r0 * LDSK + c0;
  __bf16* dB1 = dB0 + 64 * LDSK;

#if HAVE_ASYNC_LDS
  // Software pipeline: async-copy tile k+1 while WMMAs consume tile k.
  __syncthreads();                    // callers' prior LDS reads are done
  async_cp16(gA0, dA0);
  async_cp16(gA1, dA1);
  async_cp16(gB0, dB0);
  async_cp16(gB1, dB1);
  int cur = 0;
  for (int k = 0; k < K; k += BK) {
    wait_async0();                    // this wave's async stores landed
    __syncthreads();                  // tile k visible; old-buffer reads done
    int nxt = cur ^ 1;
    if (k + BK < K) {
      async_cp16(gA0 + k + BK, dA0 + nxt * ASZ);
      async_cp16(gA1 + k + BK, dA1 + nxt * ASZ);
      async_cp16(gB0 + k + BK, dB0 + nxt * BSZ);
      async_cp16(gB1 + k + BK, dB1 + nxt * BSZ);
    }
    mma_block(sA + cur * ASZ, sB + cur * BSZ, tid, acc);
    cur = nxt;
  }
#else
  // Fallback: straight-line synchronous staging (loads grouped before stores).
  for (int k = 0; k < K; k += BK) {
    u32x4 a0 = *(const u32x4*)(gA0 + k);
    u32x4 a1 = *(const u32x4*)(gA1 + k);
    u32x4 b0 = *(const u32x4*)(gB0 + k);
    u32x4 b1 = *(const u32x4*)(gB1 + k);
    if (k + BK < K) {
      __builtin_prefetch(gA0 + k + BK, 0, 1);
      __builtin_prefetch(gB0 + k + BK, 0, 1);
    }
    __syncthreads();                  // previous iteration's reads done
    *(u32x4*)dA0 = a0;
    *(u32x4*)dA1 = a1;
    *(u32x4*)dB0 = b0;
    *(u32x4*)dB1 = b1;
    __syncthreads();
    mma_block(sA, sB, tid, acc);
  }
#endif
}

__device__ __forceinline__ void zero_acc(v8f acc[2][4]) {
#pragma unroll
  for (int i = 0; i < 2; ++i)
#pragma unroll
    for (int j = 0; j < 4; ++j)
#pragma unroll
      for (int e = 0; e < 8; ++e) acc[i][j][e] = 0.0f;
}

// C/D layout: VGPR e -> M = e (lanes 0-15) / e+8 (lanes 16-31); N = lane&15.
#define EPILOGUE_FOREACH(BODY)                                            \
  {                                                                       \
    int lane = threadIdx.x & 31;                                          \
    int wave = (int)threadIdx.x >> 5;                                     \
    int mrow0 = mBase + (wave & 3) * 32;                                  \
    int ncol0 = nBase + (wave >> 2) * 64;                                 \
    _Pragma("unroll") for (int im = 0; im < 2; ++im)                      \
    _Pragma("unroll") for (int in = 0; in < 4; ++in)                      \
    _Pragma("unroll") for (int e = 0; e < 8; ++e) {                       \
      int m = mrow0 + im * 16 + ((lane & 16) ? (e + 8) : e);              \
      int n = ncol0 + in * 16 + (lane & 15);                              \
      float v = acc[im][in][e];                                           \
      BODY                                                                \
    }                                                                     \
  }

__device__ __forceinline__ float sigmoidf_(float x) {
  return 1.0f / (1.0f + __expf(-x));
}

// ---------------------------------------------------------------------------
// Conversion / transpose / init kernels
// ---------------------------------------------------------------------------
__global__ __launch_bounds__(256) void cvt_x_kernel(const float* __restrict__ in,
                                                    __bf16* __restrict__ out, int n4) {
  for (int i = blockIdx.x * blockDim.x + threadIdx.x; i < n4; i += gridDim.x * blockDim.x) {
    f32x4 v = *(const f32x4*)(in + (size_t)i * 4);
    v4bf o;
    o[0] = (__bf16)v[0]; o[1] = (__bf16)v[1]; o[2] = (__bf16)v[2]; o[3] = (__bf16)v[3];
    *(v4bf*)(out + (size_t)i * 4) = o;
  }
}

// in: [R][C] f32 row-major  ->  out: [C][R] bf16 row-major
__global__ __launch_bounds__(256) void transpose_cvt_kernel(const float* __restrict__ in,
                                                            __bf16* __restrict__ out,
                                                            int R, int C) {
  __shared__ float tile[32][33];
  int c0 = blockIdx.x * 32;
  int r0 = blockIdx.y * 32;
  int tx = threadIdx.x, ty = threadIdx.y;
  for (int j = ty; j < 32; j += 8)
    tile[j][tx] = in[(size_t)(r0 + j) * C + c0 + tx];
  __syncthreads();
  for (int j = ty; j < 32; j += 8)
    out[(size_t)(c0 + j) * R + r0 + tx] = (__bf16)tile[tx][j];
}

__global__ __launch_bounds__(256) void zero_h_kernel(float* __restrict__ hf,
                                                     __bf16* __restrict__ hb, int n) {
  for (int i = blockIdx.x * blockDim.x + threadIdx.x; i < n; i += gridDim.x * blockDim.x) {
    hf[i] = 0.0f;
    hb[i] = (__bf16)0.0f;
  }
}

// ---------------------------------------------------------------------------
// Step kernels
// ---------------------------------------------------------------------------
// r/z gates fused over N = 2H. Tiles [0..8) -> r, [8..16) -> z.
//   pre = xt@W + h@U ; g = sigmoid(pre + bias)
//   r-tiles write rh = r*h (bf16); z-tiles write z (f32).
__global__ __launch_bounds__(256) void rz_kernel(
    const __bf16* __restrict__ xt,   // [B][T*D] row view, offset to step t
    const __bf16* __restrict__ hb,   // h bf16 [B][H]
    const float* __restrict__ hf,    // h f32  [B][H]
    const __bf16* __restrict__ WrT, const __bf16* __restrict__ WzT,  // [H][D]
    const __bf16* __restrict__ UrT, const __bf16* __restrict__ UzT,  // [H][H]
    const float* __restrict__ br, const float* __restrict__ bz,
    __bf16* __restrict__ rh,         // [B][H] bf16
    float* __restrict__ zf) {        // [B][H] f32
  __shared__ __bf16 sA[2 * ASZ];
  __shared__ __bf16 sB[2 * BSZ];
  int tid = threadIdx.x;
  int mBase = blockIdx.y * BM;
  bool isR = blockIdx.x < (Hh / BN);
  int nBase = (blockIdx.x & (Hh / BN - 1)) * BN;
  const __bf16* Wt = isR ? WrT : WzT;
  const __bf16* Ut = isR ? UrT : UzT;
  const float* bias = isR ? br : bz;

  v8f acc[2][4];
  zero_acc(acc);
  gemm_tiles(xt, Tt * Dd, Wt, Dd, Dd, mBase, nBase, sA, sB, acc, tid);
  gemm_tiles(hb, Hh, Ut, Hh, Hh, mBase, nBase, sA, sB, acc, tid);

  EPILOGUE_FOREACH({
    float s = sigmoidf_(v + bias[n]);
    size_t idx = (size_t)m * Hh + n;
    if (isR) rh[idx] = (__bf16)(s * hf[idx]);
    else     zf[idx] = s;
  })
}

//   hc = relu(xt@Wh + rh@Uh + bh) ; h_new = (1-z)*hc + z*h
__global__ __launch_bounds__(256) void h_kernel(
    const __bf16* __restrict__ xt,
    const __bf16* __restrict__ rh,
    const __bf16* __restrict__ WhT,  // [H][D]
    const __bf16* __restrict__ UhT,  // [H][H]
    const float* __restrict__ bh,
    const float* __restrict__ zf,
    float* __restrict__ hf,
    __bf16* __restrict__ hb) {
  __shared__ __bf16 sA[2 * ASZ];
  __shared__ __bf16 sB[2 * BSZ];
  int tid = threadIdx.x;
  int mBase = blockIdx.y * BM;
  int nBase = blockIdx.x * BN;

  v8f acc[2][4];
  zero_acc(acc);
  gemm_tiles(xt, Tt * Dd, WhT, Dd, Dd, mBase, nBase, sA, sB, acc, tid);
  gemm_tiles(rh, Hh, UhT, Hh, Hh, mBase, nBase, sA, sB, acc, tid);

  EPILOGUE_FOREACH({
    size_t idx = (size_t)m * Hh + n;
    float hc = fmaxf(v + bh[n], 0.0f);
    float z = zf[idx];
    float hn = (1.0f - z) * hc + z * hf[idx];
    hf[idx] = hn;
    hb[idx] = (__bf16)hn;
  })
}

//   out = relu(h @ w_out + b_out)
__global__ __launch_bounds__(256) void out_kernel(
    const __bf16* __restrict__ hb,
    const __bf16* __restrict__ WoT,  // [U][H]
    const float* __restrict__ bout,
    float* __restrict__ out) {
  __shared__ __bf16 sA[2 * ASZ];
  __shared__ __bf16 sB[2 * BSZ];
  int tid = threadIdx.x;
  int mBase = blockIdx.y * BM;
  int nBase = blockIdx.x * BN;

  v8f acc[2][4];
  zero_acc(acc);
  gemm_tiles(hb, Hh, WoT, Hh, Hh, mBase, nBase, sA, sB, acc, tid);

  EPILOGUE_FOREACH({
    out[(size_t)m * Uu + n] = fmaxf(v + bout[n], 0.0f);
  })
}

// ---------------------------------------------------------------------------
// Host launcher
// ---------------------------------------------------------------------------
extern "C" void kernel_launch(void* const* d_in, const int* in_sizes, int n_in,
                              void* d_out, int out_size, void* d_ws, size_t ws_size,
                              hipStream_t stream) {
  (void)in_sizes; (void)n_in; (void)out_size; (void)ws_size;

  const float* x    = (const float*)d_in[0];
  const float* wr   = (const float*)d_in[1];
  const float* br   = (const float*)d_in[2];
  const float* wz   = (const float*)d_in[3];
  const float* bz   = (const float*)d_in[4];
  const float* wh   = (const float*)d_in[5];
  const float* bh   = (const float*)d_in[6];
  const float* Ur   = (const float*)d_in[7];
  const float* Uz   = (const float*)d_in[8];
  const float* Uh   = (const float*)d_in[9];
  const float* wout = (const float*)d_in[10];
  const float* bout = (const float*)d_in[11];
  float* out = (float*)d_out;

  // Workspace carve-up (256B aligned)
  char* base = (char*)d_ws;
  size_t off = 0;
  auto carve = [&](size_t bytes) -> void* {
    void* p = base + off;
    off = (off + bytes + 255) & ~(size_t)255;
    return p;
  };
  __bf16* x_bf   = (__bf16*)carve((size_t)Bsz * Tt * Dd * 2);
  __bf16* WrT    = (__bf16*)carve((size_t)Hh * Dd * 2);
  __bf16* WzT    = (__bf16*)carve((size_t)Hh * Dd * 2);
  __bf16* WhT    = (__bf16*)carve((size_t)Hh * Dd * 2);
  __bf16* UrT    = (__bf16*)carve((size_t)Hh * Hh * 2);
  __bf16* UzT    = (__bf16*)carve((size_t)Hh * Hh * 2);
  __bf16* UhT    = (__bf16*)carve((size_t)Hh * Hh * 2);
  __bf16* WoT    = (__bf16*)carve((size_t)Uu * Hh * 2);
  float*  h_f32  = (float*)carve((size_t)Bsz * Hh * 4);
  __bf16* h_bf   = (__bf16*)carve((size_t)Bsz * Hh * 2);
  __bf16* rh_bf  = (__bf16*)carve((size_t)Bsz * Hh * 2);
  float*  z_f32  = (float*)carve((size_t)Bsz * Hh * 4);

  // Phase 0: precision conversion + weight transposes
  int n4 = Bsz * Tt * Dd / 4;
  cvt_x_kernel<<<8192, 256, 0, stream>>>(x, x_bf, n4);

  dim3 tb(32, 8);
  transpose_cvt_kernel<<<dim3(Hh / 32, Dd / 32), tb, 0, stream>>>(wr, WrT, Dd, Hh);
  transpose_cvt_kernel<<<dim3(Hh / 32, Dd / 32), tb, 0, stream>>>(wz, WzT, Dd, Hh);
  transpose_cvt_kernel<<<dim3(Hh / 32, Dd / 32), tb, 0, stream>>>(wh, WhT, Dd, Hh);
  transpose_cvt_kernel<<<dim3(Hh / 32, Hh / 32), tb, 0, stream>>>(Ur, UrT, Hh, Hh);
  transpose_cvt_kernel<<<dim3(Hh / 32, Hh / 32), tb, 0, stream>>>(Uz, UzT, Hh, Hh);
  transpose_cvt_kernel<<<dim3(Hh / 32, Hh / 32), tb, 0, stream>>>(Uh, UhT, Hh, Hh);
  transpose_cvt_kernel<<<dim3(Uu / 32, Hh / 32), tb, 0, stream>>>(wout, WoT, Hh, Uu);

  zero_h_kernel<<<2048, 256, 0, stream>>>(h_f32, h_bf, Bsz * Hh);

  // Phase 1: T sequential GRU steps (WMMA bf16 GEMMs)
  dim3 gridRZ(2 * Hh / BN, Bsz / BM);  // 16 x 64
  dim3 gridH(Hh / BN, Bsz / BM);       //  8 x 64
  for (int t = 0; t < Tt; ++t) {
    const __bf16* xt = x_bf + (size_t)t * Dd;  // row stride T*D
    rz_kernel<<<gridRZ, 256, 0, stream>>>(xt, h_bf, h_f32, WrT, WzT, UrT, UzT,
                                          br, bz, rh_bf, z_f32);
    h_kernel<<<gridH, 256, 0, stream>>>(xt, rh_bf, WhT, UhT, bh, z_f32, h_f32, h_bf);
  }

  // Phase 2: output projection
  dim3 gridO(Uu / BN, Bsz / BM);       // 16 x 64
  out_kernel<<<gridO, 256, 0, stream>>>(h_bf, WoT, bout, out);
}
